// LSTMModel_39221641347102
// MI455X (gfx1250) — matrix operand order repaired and checked by reading, not compiled, not run on hardware
//
#include <hip/hip_runtime.h>
#include <stdint.h>

typedef __attribute__((ext_vector_type(16))) __bf16 v16bf;
typedef __attribute__((ext_vector_type(8)))  float  v8f;

#define LSTM_T 512
#define LSTM_B 2048
#define LSTM_H 16
#define LSTM_L 8
#define NTILE  (LSTM_B / 16)     // 128 batch tiles
#define CHUNK  16                // timesteps per pipeline chunk (power of 2)

// ---------- helpers ----------
__device__ __forceinline__ unsigned pk_bf16(float a, float b) {
    union { float f; unsigned u; } ua, ub;
    ua.f = a; ub.f = b;
    unsigned ha = (ua.u + 0x7FFFu + ((ua.u >> 16) & 1u)) >> 16;   // RNE
    unsigned hb = (ub.u + 0x7FFFu + ((ub.u >> 16) & 1u)) >> 16;
    return (hb << 16) | (ha & 0xFFFFu);
}
__device__ __forceinline__ float bf_lo(unsigned u) {
    union { unsigned u; float f; } c; c.u = u << 16; return c.f;
}
__device__ __forceinline__ float bf_hi(unsigned u) {
    union { unsigned u; float f; } c; c.u = u & 0xFFFF0000u; return c.f;
}

#if __has_builtin(__builtin_amdgcn_tanhf)
// gfx1250 hardware v_tanh_f32 (confirmed in round-2 disasm)
__device__ __forceinline__ float ftanh(float x) { return __builtin_amdgcn_tanhf(x); }
// sigmoid with pre-scaled bias bh = 0.5*b: sigmoid(x+b) = 0.5*tanh(0.5x + bh) + 0.5
__device__ __forceinline__ float fsigmoid_b(float x, float bh) {
    return fmaf(__builtin_amdgcn_tanhf(fmaf(x, 0.5f, bh)), 0.5f, 0.5f);
}
__device__ __forceinline__ float fsigmoid(float x) {
    return fmaf(__builtin_amdgcn_tanhf(0.5f * x), 0.5f, 0.5f);
}
#else
__device__ __forceinline__ float ftanh(float x) {
    float e = __builtin_amdgcn_exp2f(-2.885390082f * x);
    return 2.0f * __builtin_amdgcn_rcpf(1.0f + e) - 1.0f;
}
__device__ __forceinline__ float fsigmoid_b(float x, float bh) {
    float e = __builtin_amdgcn_exp2f(-2.885390082f * fmaf(x, 0.5f, bh));
    return __builtin_amdgcn_rcpf(1.0f + e);
}
__device__ __forceinline__ float fsigmoid(float x) {
    float e = __builtin_amdgcn_exp2f(-1.442695041f * x);
    return __builtin_amdgcn_rcpf(1.0f + e);
}
#endif

union BU { unsigned u[8]; v16bf v; };

// ---------- fp32 -> packed bf16 conversion (first-layer input) ----------
__global__ __launch_bounds__(256)
void cvt_f32_bf16pk(const float2* __restrict__ in, unsigned* __restrict__ out, int n) {
    int i = blockIdx.x * blockDim.x + threadIdx.x;
    if (i < n) { float2 v = in[i]; out[i] = pk_bf16(v.x, v.y); }
}

// ---------- zero the pipeline flags (every launch; graph-replay safe) ----------
__global__ __launch_bounds__(256)
void zero_flags(unsigned* __restrict__ flags, int n) {
    int i = blockIdx.x * blockDim.x + threadIdx.x;
    if (i < n) flags[i] = 0u;
}

// ---------- fused pipelined LSTM: all 8 layers concurrent ----------
// blockIdx.x = batch tile (0..127), blockIdx.y = layer (0..7).
// One wave owns 16 batch elements of one layer for all 512 timesteps.
// Layer l+1 chases layer l through chunk-completion flags (release/acquire, agent scope).
// Block order is layer-major, so each block depends only on lower-indexed blocks.
__global__ __launch_bounds__(32)
void lstm_pipe_kernel(unsigned* __restrict__ b0, unsigned* __restrict__ b1,
                      unsigned* __restrict__ flags,
                      const float* __restrict__ WihAll,   // [8][64][16]
                      const float* __restrict__ WhhAll,   // [8][64][16]
                      const float* __restrict__ bihAll,   // [8][64]
                      const float* __restrict__ bhhAll)   // [8][64]
{
    const int lane  = threadIdx.x;
    const int bloc  = lane & 15;     // batch within tile (= N, = W row within tile)
    const int half  = lane >> 4;     // lane half
    const int tile  = blockIdx.x;
    const int layer = blockIdx.y;
    const int bglob = tile * 16 + bloc;

    const unsigned* xin  = (layer & 1) ? b1 : b0;
    unsigned*       xout = (layer & 1) ? b0 : b1;
    const float* Wih = WihAll + (size_t)layer * 64 * LSTM_H;
    const float* Whh = WhhAll + (size_t)layer * 64 * LSTM_H;
    const float* bih = bihAll + (size_t)layer * 64;
    const float* bhh = bhhAll + (size_t)layer * 64;

    // ---- preload A (weights) in contiguous k = 16*half + s order ----
    // VGPR v: K = 16*half + 2v  (half 0: Wih part, K 0..15; half 1: Whh part, K 16..31)
    BU Abu[4];
    float bsc[4][8];                 // bias folded into activations (i/f/o pre-scaled by 0.5)
    const int koff = half * 8;
#pragma unroll
    for (int g = 0; g < 4; ++g) {
        const float* wi = Wih + (g * 16 + bloc) * LSTM_H;
        const float* wh = Whh + (g * 16 + bloc) * LSTM_H;
#pragma unroll
        for (int v = 0; v < 4; ++v)
            Abu[g].u[v] = pk_bf16((half ? wh : wi)[2 * v], (half ? wh : wi)[2 * v + 1]);
#pragma unroll
        for (int v = 0; v < 4; ++v)
            Abu[g].u[4 + v] = pk_bf16((half ? wh : wi)[2 * v + 8], (half ? wh : wi)[2 * v + 9]);
#pragma unroll
        for (int r = 0; r < 8; ++r) {
            int jb = g * 16 + r + half * 8;     // D layout: M = r + half*8
            float bb = bih[jb] + bhh[jb];
            bsc[g][r] = (g == 2) ? bb : 0.5f * bb;  // g-gate raw, sigmoids half-scaled
        }
    }
    const v8f zC = {};               // zero C: loop-invariant, no per-iteration copies

    // ---- state: lane holds c/h for batch=bloc, hidden = r + half*8 ----
    float cst[8];
    unsigned p[4];
#pragma unroll
    for (int r = 0; r < 8; ++r) cst[r] = 0.0f;
#pragma unroll
    for (int v = 0; v < 4; ++v) p[v] = 0u;

    const unsigned* xbase = xin + (size_t)bglob * LSTM_T * (LSTM_H / 2);
    unsigned* obase = xout + (size_t)bglob * LSTM_T * (LSTM_H / 2) + half * 4;

    unsigned* myflag = flags + layer * NTILE + tile;
    const unsigned* upflag = flags + (layer - 1) * NTILE + tile;

    // wait for chunk 0 from the upstream layer (layer 0 input came from cvt kernel)
    if (layer > 0) {
        while (__hip_atomic_load(upflag, __ATOMIC_ACQUIRE, __HIP_MEMORY_SCOPE_AGENT) < 1u)
            __builtin_amdgcn_s_sleep(2);
    }

    uint4 xa0 = *(const uint4*)(xbase + 0);
    uint4 xa1 = *(const uint4*)(xbase + 4);

    for (int t = 0; t < LSTM_T; ++t) {
        const int tp1 = t + 1;
        // before prefetching into the next chunk, wait until upstream produced it
        if (layer > 0 && tp1 < LSTM_T && (tp1 & (CHUNK - 1)) == 0) {
            unsigned need = (unsigned)(tp1 / CHUNK) + 1u;
            while (__hip_atomic_load(upflag, __ATOMIC_ACQUIRE, __HIP_MEMORY_SCOPE_AGENT) < need)
                __builtin_amdgcn_s_sleep(2);
        }
        const int tn = (tp1 < LSTM_T) ? tp1 : t;
        uint4 xb0 = *(const uint4*)(xbase + (size_t)tn * 8 + 0);
        uint4 xb1 = *(const uint4*)(xbase + (size_t)tn * 8 + 4);

        // ---- build B operand: lanes 0-15 K=0..15 (x), lanes 16-31 K=16..31 (h) ----
        unsigned xp[8] = {xa0.x, xa0.y, xa0.z, xa0.w, xa1.x, xa1.y, xa1.z, xa1.w};
        BU Bbu;
#pragma unroll
        for (int v = 0; v < 4; ++v) {
            // high lanes need hidden (2v,2v+1): lives in low lane bloc's p[v]
            unsigned hv = (unsigned)__builtin_amdgcn_ds_bpermute(bloc << 2, (int)p[v]);
            Bbu.u[v] = half ? hv : xp[v];
        }
#pragma unroll
        for (int v = 4; v < 8; ++v)
            Bbu.u[v] = half ? p[v - 4] : xp[v];  // hidden (2v,2v+1) already local

        // ---- 4 WMMAs: i, f, g, o gate tiles (C = 0; bias folded into activations) ----
        v8f di = __builtin_amdgcn_wmma_f32_16x16x32_bf16(false, Abu[0].v, false, Bbu.v, (short)0, zC, false, false);
        v8f df = __builtin_amdgcn_wmma_f32_16x16x32_bf16(false, Abu[1].v, false, Bbu.v, (short)0, zC, false, false);
        v8f dg = __builtin_amdgcn_wmma_f32_16x16x32_bf16(false, Abu[2].v, false, Bbu.v, (short)0, zC, false, false);
        v8f dd = __builtin_amdgcn_wmma_f32_16x16x32_bf16(false, Abu[3].v, false, Bbu.v, (short)0, zC, false, false);

        // ---- lane-local elementwise state update (bias folded in) ----
        float hn[8];
#pragma unroll
        for (int r = 0; r < 8; ++r) {
            float iv = fsigmoid_b(di[r], bsc[0][r]);
            float fv = fsigmoid_b(df[r], bsc[1][r]);
            float gv = ftanh(dg[r] + bsc[2][r]);
            float ov = fsigmoid_b(dd[r], bsc[3][r]);
            cst[r] = fv * cst[r] + iv * gv;
            hn[r] = ov * ftanh(cst[r]);
        }
#pragma unroll
        for (int v = 0; v < 4; ++v) p[v] = pk_bf16(hn[2 * v], hn[2 * v + 1]);

        // store h_t (this half's 8 hidden values = 16 bytes)
        uint4 st; st.x = p[0]; st.y = p[1]; st.z = p[2]; st.w = p[3];
        *(uint4*)(obase + (size_t)t * 8) = st;

        // publish completed chunk to the downstream layer
        if ((tp1 & (CHUNK - 1)) == 0) {
            unsigned done = (unsigned)(tp1 / CHUNK);
            if (lane == 0)
                __hip_atomic_store(myflag, done, __ATOMIC_RELEASE, __HIP_MEMORY_SCOPE_AGENT);
        }

        xa0 = xb0; xa1 = xb1;
    }
}

// ---------- head: pos = h W_p^T + b_p ; ori = tanh(h W_o^T + b_o) ----------
__global__ __launch_bounds__(256)
void head_kernel(const unsigned* __restrict__ hbuf,
                 const float* __restrict__ Wp, const float* __restrict__ bp,
                 const float* __restrict__ Wo, const float* __restrict__ bo,
                 float* __restrict__ out)
{
    int b = blockIdx.x * blockDim.x + threadIdx.x;
    if (b >= LSTM_B) return;
    const unsigned* hp = hbuf + ((size_t)b * LSTM_T + (LSTM_T - 1)) * (LSTM_H / 2);
    float h[16];
#pragma unroll
    for (int v = 0; v < 8; ++v) {
        unsigned u = hp[v];
        h[2 * v] = bf_lo(u);
        h[2 * v + 1] = bf_hi(u);
    }
#pragma unroll
    for (int k = 0; k < 3; ++k) {
        float acc = bp[k];
#pragma unroll
        for (int j = 0; j < 16; ++j) acc += h[j] * Wp[k * 16 + j];
        out[b * 6 + k] = acc;
    }
#pragma unroll
    for (int k = 0; k < 3; ++k) {
        float acc = bo[k];
#pragma unroll
        for (int j = 0; j < 16; ++j) acc += h[j] * Wo[k * 16 + j];
        out[b * 6 + 3 + k] = ftanh(acc);
    }
}

// ---------- launch ----------
extern "C" void kernel_launch(void* const* d_in, const int* in_sizes, int n_in,
                              void* d_out, int out_size, void* d_ws, size_t ws_size,
                              hipStream_t stream) {
    const float* x    = (const float*)d_in[0];   // [2048][512][16]
    const float* W_ih = (const float*)d_in[1];   // [8][64][16]
    const float* W_hh = (const float*)d_in[2];   // [8][64][16]
    const float* b_ih = (const float*)d_in[3];   // [8][64]
    const float* b_hh = (const float*)d_in[4];   // [8][64]
    const float* Wp   = (const float*)d_in[5];   // [3][16]
    const float* bp   = (const float*)d_in[6];   // [3]
    const float* Wo   = (const float*)d_in[7];   // [3][16]
    const float* bo   = (const float*)d_in[8];   // [3]
    float* out        = (float*)d_out;           // [2048][6]

    const size_t buf_elems = (size_t)LSTM_B * LSTM_T * (LSTM_H / 2); // uints (2x bf16)
    unsigned* buf0  = (unsigned*)d_ws;
    unsigned* buf1  = buf0 + buf_elems;
    unsigned* flags = buf1 + buf_elems;          // [8][128] chunk counters
    const int nflags = LSTM_L * NTILE;

    // 1) zero pipeline flags + convert x fp32 -> packed bf16 into buf0
    zero_flags<<<(nflags + 255) / 256, 256, 0, stream>>>(flags, nflags);
    {
        int n = (int)buf_elems;                  // 8,388,608 pairs
        cvt_f32_bf16pk<<<(n + 255) / 256, 256, 0, stream>>>((const float2*)x, buf0, n);
    }

    // 2) all 8 layers pipelined in one launch; layers chase each other through flags.
    //    layer-major block order guarantees forward progress under partial residency.
    {
        dim3 grid(NTILE, LSTM_L, 1);
        lstm_pipe_kernel<<<grid, 32, 0, stream>>>(buf0, buf1, flags,
                                                  W_ih, W_hh, b_ih, b_hh);
    }
    // after 8 layers the output sits in buf0 (layer 7 writes buf0)

    // 3) head
    head_kernel<<<(LSTM_B + 255) / 256, 256, 0, stream>>>(buf0, Wp, bp, Wo, bo, out);
}
